// Model_9380208574536
// MI455X (gfx1250) — matrix operand order repaired and checked by reading, not compiled, not run on hardware
//
#include <hip/hip_runtime.h>

typedef unsigned short u16;
typedef unsigned int   u32;
typedef __attribute__((ext_vector_type(16))) __bf16 bf16x16;
typedef __attribute__((ext_vector_type(8)))  float  floatx8;
typedef __attribute__((ext_vector_type(4)))  u32    u32x4;
typedef __attribute__((ext_vector_type(4)))  float  f32x4;
typedef __attribute__((ext_vector_type(4)))  u16    u16x4;

union Frag { u32x4 q[2]; bf16x16 v; };   // 32 bytes: 16 bf16 per lane

#define HDIM   1024
#define VDIM   50257
#define MTOT   4096            // B*S tokens
#define KFRAGS 32              // HDIM / 32
#define NTILES 3142            // ceil(VDIM/16)
#define NCHUNK 4               // vocab-range split across blockIdx.y
#define MTILE  64              // tokens per block
#define NSUB   4               // MTILE / 16 WMMA row-tiles
#define MBLOCKS (MTOT / MTILE) // 64

__device__ __forceinline__ u16 f2bf(float f) {        // fp32 -> bf16 RNE
  u32 u = __float_as_uint(f);
  u += 0x7FFFu + ((u >> 16) & 1u);
  return (u16)(u >> 16);
}
__device__ __forceinline__ float bf2f(u16 h) {
  return __uint_as_float(((u32)h) << 16);
}

// ---------------------------------------------------------------------------
// Kernel 1: split W (fp32) into bf16 hi/lo planes (one pass, HBM-bound)
// ---------------------------------------------------------------------------
__global__ __launch_bounds__(256) void convert_w_kernel(
    const float* __restrict__ W, u16* __restrict__ Whi, u16* __restrict__ Wlo,
    int n4) {
  int i = blockIdx.x * 256 + threadIdx.x;
  int stride = gridDim.x * 256;
  for (; i < n4; i += stride) {
    f32x4 x = ((const f32x4*)W)[i];
    u16x4 h, l;
#pragma unroll
    for (int c = 0; c < 4; ++c) {
      u16 hb = f2bf(x[c]);
      h[c] = hb;
      l[c] = f2bf(x[c] - bf2f(hb));
    }
    ((u16x4*)Whi)[i] = h;
    ((u16x4*)Wlo)[i] = l;
  }
}

// ---------------------------------------------------------------------------
// Kernel 2: fused GEMM (bf16-split WMMA) + online softmax statistics
//   grid = (MBLOCKS, NCHUNK), block = 256 (8 wave32)
//   stats[NCHUNK][MTOT][3] = {row_max, row_sumexp, target_logit}
// ---------------------------------------------------------------------------
__global__ __launch_bounds__(256) void lmhead_ce_kernel(
    const float* __restrict__ hidden,
    const u16*   __restrict__ Whi,
    const u16*   __restrict__ Wlo,
    const float* __restrict__ bias,
    const int*   __restrict__ targets,
    float*       __restrict__ stats) {
  // A fragments stored in HW A-layout: addr = sub*16384 + kf*512 + lane*16 + j
  __shared__ u16 a_hi[NSUB * KFRAGS * 512];
  __shared__ u16 a_lo[NSUB * KFRAGS * 512];
  __shared__ float red_max[8][MTILE];
  __shared__ float red_sum[8][MTILE];
  __shared__ float red_tgt[8][MTILE];
  __shared__ int   tgt_s[MTILE];

  const int tid  = threadIdx.x;
  const int wave = tid >> 5;
  const int lane = tid & 31;
  const int m0   = blockIdx.x * MTILE;
  const int chunk = blockIdx.y;
  const float NEG_INF = -__builtin_inff();

  if (tid < MTILE) tgt_s[tid] = targets[m0 + tid];

  // ---- fill A fragments (bf16 hi/lo), exact ISA 16-bit A 16x32 layout ----
  for (int idx = tid; idx < MTILE * HDIM; idx += 256) {
    int ml = idx >> 10;            // local row 0..63
    int k  = idx & (HDIM - 1);
    float x = hidden[(size_t)(m0 + ml) * HDIM + k];
    u16 hb = f2bf(x);
    u16 lb = f2bf(x - bf2f(hb));
    int st  = ml >> 4, m16 = ml & 15;
    int f   = k >> 5,  kp  = k & 31;
    int l   = m16 + (((kp >> 3) & 1) << 4);      // K 8-15,24-31 -> lanes 16-31
    int j   = (kp & 7) + ((kp & 16) >> 1);       // elem index in v16bf
    int addr = st * (KFRAGS * 512) + f * 512 + l * 16 + j;
    a_hi[addr] = hb;
    a_lo[addr] = lb;
  }
  __syncthreads();

  const int colg   = lane & 15;          // C/D column, B column (vocab)
  const int khalf  = (lane >> 4) * 8;    // B frag: lanes 16-31 hold K 8-15,24-31
  const int rowoff = (lane >> 4) * 8;    // C frag: lanes 16-31 hold rows 8-15

  // Per-LANE online softmax state over the columns this lane computes.
  float run_max[NSUB][8], run_sum[NSUB][8], tgtl[NSUB][8];
#pragma unroll
  for (int s = 0; s < NSUB; ++s)
#pragma unroll
    for (int r = 0; r < 8; ++r) {
      run_max[s][r] = NEG_INF; run_sum[s][r] = 0.f; tgtl[s][r] = NEG_INF;
    }

  const int chunkSz = (NTILES + NCHUNK - 1) / NCHUNK;
  const int ntBase  = chunk * chunkSz;
  const int ntEnd   = (ntBase + chunkSz < NTILES) ? ntBase + chunkSz : NTILES;

#pragma clang loop unroll(disable)
  for (int nt = ntBase + wave; nt < ntEnd; nt += 8) {
    int n0 = nt * 16;
    int v  = n0 + colg;
    bool vvalid = v < VDIM;
    int vc = vvalid ? v : (VDIM - 1);
    const u16* wh = Whi + (size_t)vc * HDIM + khalf;
    const u16* wl = Wlo + (size_t)vc * HDIM + khalf;

    // prefetch next tile's W rows (global_prefetch_b8)
    int ntn = nt + 8;
    if (ntn < ntEnd) {
      int vn = ntn * 16 + colg; if (vn >= VDIM) vn = VDIM - 1;
      __builtin_prefetch(Whi + (size_t)vn * HDIM, 0, 3);
      __builtin_prefetch(Wlo + (size_t)vn * HDIM, 0, 3);
    }

    floatx8 acc[NSUB] = {};
    // Keep live ranges short: do NOT unroll the K loop (spills otherwise).
#pragma clang loop unroll(disable)
    for (int kf = 0; kf < KFRAGS; ++kf) {
      Frag bh, bl;
      bh.q[0] = *(const u32x4*)(wh + kf * 32);
      bh.q[1] = *(const u32x4*)(wh + kf * 32 + 16);
      bl.q[0] = *(const u32x4*)(wl + kf * 32);
      bl.q[1] = *(const u32x4*)(wl + kf * 32 + 16);
#pragma unroll
      for (int s = 0; s < NSUB; ++s) {
        Frag ah, al;
        const u16* ap  = &a_hi[s * (KFRAGS * 512) + kf * 512 + lane * 16];
        const u16* alp = &a_lo[s * (KFRAGS * 512) + kf * 512 + lane * 16];
        ah.q[0] = *(const u32x4*)(ap);
        ah.q[1] = *(const u32x4*)(ap + 8);
        al.q[0] = *(const u32x4*)(alp);
        al.q[1] = *(const u32x4*)(alp + 8);
        // fp32 emulation: hi*hi + hi*lo + lo*hi
        acc[s] = __builtin_amdgcn_wmma_f32_16x16x32_bf16(
            false, ah.v, false, bh.v, (short)0, acc[s], false, false);
        acc[s] = __builtin_amdgcn_wmma_f32_16x16x32_bf16(
            false, ah.v, false, bl.v, (short)0, acc[s], false, false);
        acc[s] = __builtin_amdgcn_wmma_f32_16x16x32_bf16(
            false, al.v, false, bh.v, (short)0, acc[s], false, false);
      }
    }

    // Per-lane online update (no cross-lane work per tile).
    float bval = bias[vc];
#pragma unroll
    for (int s = 0; s < NSUB; ++s) {
#pragma unroll
      for (int r = 0; r < 8; ++r) {
        if (vvalid) {
          float lg = acc[s][r] + bval;
          int row = s * 16 + rowoff + r;
          if (tgt_s[row] == v) tgtl[s][r] = lg;
          float nm = fmaxf(run_max[s][r], lg);
          run_sum[s][r] = run_sum[s][r] * __expf(run_max[s][r] - nm)
                        + __expf(lg - nm);
          run_max[s][r] = nm;
        }
      }
    }
  }

  // One cross-lane online merge (16-lane groups), done once.
#pragma unroll
  for (int s = 0; s < NSUB; ++s)
#pragma unroll
    for (int r = 0; r < 8; ++r) {
#pragma unroll
      for (int d = 1; d < 16; d <<= 1) {
        float om = __shfl_xor(run_max[s][r], d, 32);
        float os = __shfl_xor(run_sum[s][r], d, 32);
        float nm = fmaxf(run_max[s][r], om);
        run_sum[s][r] = run_sum[s][r] * __expf(run_max[s][r] - nm)
                      + os * __expf(om - nm);
        run_max[s][r] = nm;
        tgtl[s][r] = fmaxf(tgtl[s][r], __shfl_xor(tgtl[s][r], d, 32));
      }
    }

  if ((lane & 15) == 0) {
#pragma unroll
    for (int s = 0; s < NSUB; ++s)
      for (int r = 0; r < 8; ++r) {
        int row = s * 16 + rowoff + r;
        red_max[wave][row] = run_max[s][r];
        red_sum[wave][row] = run_sum[s][r];
        red_tgt[wave][row] = tgtl[s][r];
      }
  }
  __syncthreads();

  if (tid < MTILE) {
    float M = NEG_INF, S = 0.f, T = NEG_INF;
    for (int w = 0; w < 8; ++w) {
      float m_ = red_max[w][tid], s_ = red_sum[w][tid], t_ = red_tgt[w][tid];
      float nm = fmaxf(M, m_);
      S = S * __expf(M - nm) + s_ * __expf(m_ - nm);
      M = nm;
      T = fmaxf(T, t_);
    }
    float* p = stats + ((size_t)chunk * MTOT + (size_t)(m0 + tid)) * 3;
    p[0] = M; p[1] = S; p[2] = T;
  }
}

// ---------------------------------------------------------------------------
// Kernel 3: merge vocab chunks per token, compute mean NLL
// ---------------------------------------------------------------------------
__global__ __launch_bounds__(256) void ce_finalize_kernel(
    const float* __restrict__ stats, const int* __restrict__ targets,
    float* __restrict__ out) {
  __shared__ float ssum[256], scnt[256];
  const float NEG_INF = -__builtin_inff();
  float ls = 0.f, lc = 0.f;
  for (int tok = threadIdx.x; tok < MTOT; tok += 256) {
    float M = NEG_INF, S = 0.f, T = NEG_INF;
    for (int c = 0; c < NCHUNK; ++c) {
      const float* p = stats + ((size_t)c * MTOT + tok) * 3;
      float m_ = p[0], s_ = p[1], t_ = p[2];
      float nm = fmaxf(M, m_);
      S = S * __expf(M - nm) + s_ * __expf(m_ - nm);
      M = nm;
      T = fmaxf(T, t_);
    }
    int t = targets[tok];
    if (t >= 0) { ls += logf(S) + M - T; lc += 1.f; }
  }
  ssum[threadIdx.x] = ls; scnt[threadIdx.x] = lc;
  __syncthreads();
  for (int off = 128; off > 0; off >>= 1) {
    if (threadIdx.x < off) {
      ssum[threadIdx.x] += ssum[threadIdx.x + off];
      scnt[threadIdx.x] += scnt[threadIdx.x + off];
    }
    __syncthreads();
  }
  if (threadIdx.x == 0) out[0] = ssum[0] / fmaxf(scnt[0], 1.f);
}

// ---------------------------------------------------------------------------
extern "C" void kernel_launch(void* const* d_in, const int* in_sizes, int n_in,
                              void* d_out, int out_size, void* d_ws,
                              size_t ws_size, hipStream_t stream) {
  const float* hidden  = (const float*)d_in[0];
  const float* W       = (const float*)d_in[1];
  const float* bias    = (const float*)d_in[2];
  const int*   targets = (const int*)d_in[3];
  float* out = (float*)d_out;

  // workspace: Whi | Wlo | stats  (~206 MB + 192 KB)
  char* ws = (char*)d_ws;
  size_t plane_bytes = (size_t)VDIM * HDIM * sizeof(u16);
  u16* Whi = (u16*)ws;
  u16* Wlo = (u16*)(ws + plane_bytes);
  float* stats = (float*)(ws + 2 * plane_bytes);

  convert_w_kernel<<<2048, 256, 0, stream>>>(W, Whi, Wlo, (VDIM * HDIM) / 4);

  dim3 grid(MBLOCKS, NCHUNK);
  lmhead_ce_kernel<<<grid, 256, 0, stream>>>(hidden, Whi, Wlo, bias, targets,
                                             stats);

  ce_finalize_kernel<<<1, 256, 0, stream>>>(stats, targets, out);
}
